// VectorQuantizer_55628416418113
// MI455X (gfx1250) — matrix operand order repaired and checked by reading, not compiled, not run on hardware
//
#include <hip/hip_runtime.h>
#include <hip/hip_bf16.h>
#include <cstdint>

typedef __attribute__((ext_vector_type(16))) _Float16 v16h;
typedef __attribute__((ext_vector_type(8)))  _Float16 v8h;
typedef __attribute__((ext_vector_type(2)))  _Float16 v2h;
typedef __attribute__((ext_vector_type(8)))  float    v8f;

#define PAD 88            // row pad (halves): 176B = 11*16 -> 16B aligned, 44-bank rotation (conflict-free for 16 consecutive rows)
#define N_CODES 1024
#define STAGE_CODES 32    // codes per double-buffered stage
#define N_STAGES (N_CODES / STAGE_CODES)

// ---------------- Kernel 1: codebook f32 -> f16 + ||e||^2 ----------------
__global__ void vq_prep(const float* __restrict__ emb,
                        _Float16* __restrict__ e16,
                        float* __restrict__ enorm) {
    __shared__ float red[64];
    const int k = blockIdx.x;      // code 0..1023
    const int d = threadIdx.x;     // dim  0..63
    float v = emb[k * 64 + d];
    e16[k * 64 + d] = (_Float16)v;
    red[d] = v * v;
    __syncthreads();
    if (d == 0) {
        float s = 0.f;
        #pragma unroll
        for (int i = 0; i < 64; ++i) s += red[i];
        enorm[k] = s;
    }
}

// Issue one wave-parallel async copy of a 32-code stage (4KB) into an LDS buffer.
// 256 chunks of 16B, one per thread; tracked on ASYNCcnt, no VGPR data round-trip.
__device__ __forceinline__ void issue_stage_async(const _Float16* __restrict__ e16,
                                                  _Float16 (*EhBuf)[PAD], int s, int t) {
    const int code = t >> 3;   // 0..31
    const int part = t & 7;    // 0..7 (16B chunks of a 128B row)
    const _Float16* g = e16 + (s * STAGE_CODES + code) * 64 + part * 8;
    const unsigned lds = (unsigned)(uintptr_t)(&EhBuf[code][part * 8]);  // low 32 bits = LDS byte address
    asm volatile("global_load_async_to_lds_b128 %0, %1, off"
                 :: "v"(lds), "v"(g)
                 : "memory");
}

// ---------------- Kernel 2: fused distance+argmin+gather+partial-loss ----------------
// grid = 512 (b in 0..31, h-slab in 0..15), block = 256 (8 wave32)
__global__ __launch_bounds__(256)
void vq_main(const float* __restrict__ z,
             const float* __restrict__ emb,
             const _Float16* __restrict__ e16,
             const float* __restrict__ enorm,
             float* __restrict__ out,
             float* __restrict__ partials) {
    __shared__ _Float16 Ah[256][PAD];               // z rows (f16): row = h'*64+w, col = d
    __shared__ _Float16 Eh[2][STAGE_CODES][PAD];    // double-buffered codebook stages (f16)
    __shared__ float    enLds[N_CODES];
    __shared__ int      idxLds[256];
    __shared__ float    redLds[8];

    const int t  = threadIdx.x;
    const int b  = blockIdx.x >> 4;
    const int h0 = (blockIdx.x & 15) << 2;

    const float* zb = z   + (b * 64 * 64 * 64) + h0 * 64;   // + d*4096 + t is coalesced
    float*       ob = out + (b * 64 * 64 * 64) + h0 * 64;

    // prefetch stage 0 of the codebook into buffer 0 while we stage A
    issue_stage_async(e16, Eh[0], 0, t);

    // ---- stage A tile: z[b][d][h0..h0+3][*] -> Ah[t][d] (f16) ----
    #pragma unroll
    for (int d = 0; d < 64; d += 2) {
        float f0 = zb[d * 4096 + t];
        float f1 = zb[(d + 1) * 4096 + t];
        v2h p; p[0] = (_Float16)f0; p[1] = (_Float16)f1;
        *(v2h*)(&Ah[t][d]) = p;
    }
    #pragma unroll
    for (int i = 0; i < 4; ++i) enLds[t + i * 256] = enorm[t + i * 256];
    __syncthreads();

    const int lane    = t & 31;
    const int wv      = t >> 5;        // wave 0..7
    const int nloc    = lane & 15;
    const int hi      = lane >> 4;
    const int rowbase = wv * 32;       // this wave owns output rows rowbase..rowbase+31

    // ---- A fragments in registers for the whole K loop (ISA 16-bit A 16x32 layout) ----
    v16h afrag[2][2];
    #pragma unroll
    for (int mt = 0; mt < 2; ++mt) {
        const int m = rowbase + mt * 16 + nloc;
        #pragma unroll
        for (int kh = 0; kh < 2; ++kh) {
            const int kb = kh * 32 + hi * 8;              // VGPR0-3: K=kb..kb+7, VGPR4-7: K=kb+16..kb+23
            v8h lo = *(const v8h*)(&Ah[m][kb]);
            v8h hh = *(const v8h*)(&Ah[m][kb + 16]);
            v16h a;
            #pragma unroll
            for (int i = 0; i < 8; ++i) { a[i] = lo[i]; a[i + 8] = hh[i]; }
            afrag[mt][kh] = a;
        }
    }

    float    bestd[2][8];
    unsigned bestn[2][8];
    #pragma unroll
    for (int mt = 0; mt < 2; ++mt)
        #pragma unroll
        for (int r = 0; r < 8; ++r) { bestd[mt][r] = 3.402823466e+38f; bestn[mt][r] = 0u; }

    // ---- K loop: 32 double-buffered stages of 32 codes ----
    for (int s = 0; s < N_STAGES; ++s) {
        if (s < N_STAGES - 1) {
            issue_stage_async(e16, Eh[(s + 1) & 1], s + 1, t);   // overlap next-stage copy with this stage's WMMAs
            asm volatile("s_wait_asynccnt 0x1" ::: "memory");    // stage s landed (in-order completion)
        } else {
            asm volatile("s_wait_asynccnt 0x0" ::: "memory");
        }
        __syncthreads();
        const int buf = s & 1;

        #pragma unroll
        for (int cc = 0; cc < 2; ++cc) {
            const int cb  = s * STAGE_CODES + cc * 16;   // global code base of this chunk
            const int row = cc * 16 + nloc;              // this lane's code row in Eh
            const float en = enLds[cb + nloc];

            // B fragments (dense 16-bit B 32x16 layout: lane holds dims kb..kb+7 and kb+16..kb+23 of one code)
            v16h bfrag[2];
            #pragma unroll
            for (int kh = 0; kh < 2; ++kh) {
                const int kb = kh * 32 + hi * 8;
                v8h lo = *(const v8h*)(&Eh[buf][row][kb]);
                v8h hh = *(const v8h*)(&Eh[buf][row][kb + 16]);
                v16h bb;
                #pragma unroll
                for (int i = 0; i < 8; ++i) { bb[i] = lo[i]; bb[i + 8] = hh[i]; }
                bfrag[kh] = bb;
            }

            #pragma unroll
            for (int mt = 0; mt < 2; ++mt) {
                v8f acc = {};
                acc = __builtin_amdgcn_wmma_f32_16x16x32_f16(false, afrag[mt][0], false, bfrag[0],
                                                             (short)0, acc, false, false);
                acc = __builtin_amdgcn_wmma_f32_16x16x32_f16(false, afrag[mt][1], false, bfrag[1],
                                                             (short)0, acc, false, false);
                #pragma unroll
                for (int r = 0; r < 8; ++r) {
                    const float dist = __builtin_fmaf(-2.0f, acc[r], en);  // ||e||^2 - 2 z.e
                    const bool lt = dist < bestd[mt][r];
                    bestd[mt][r] = lt ? dist : bestd[mt][r];
                    bestn[mt][r] = lt ? (unsigned)(cb + nloc) : bestn[mt][r];
                }
            }
        }
        __syncthreads();   // all waves done reading buf before it is re-issued at s+2
    }

    // ---- cross-lane argmin: u64 (ordered-float<<32 | idx) min => min dist, tie -> min idx ----
    #pragma unroll
    for (int mt = 0; mt < 2; ++mt) {
        #pragma unroll
        for (int r = 0; r < 8; ++r) {
            unsigned u = __float_as_uint(bestd[mt][r]);
            u = (u & 0x80000000u) ? ~u : (u | 0x80000000u);
            unsigned long long key = ((unsigned long long)u << 32) | (unsigned long long)bestn[mt][r];
            #pragma unroll
            for (int mask = 1; mask <= 8; mask <<= 1) {
                unsigned long long o = __shfl_xor(key, mask, 32);
                key = (o < key) ? o : key;
            }
            if (nloc == 0)
                idxLds[rowbase + mt * 16 + hi * 8 + r] = (int)(unsigned)key;
        }
    }
    __syncthreads();

    // ---- epilogue: gather f32 code, write transposed output, accumulate SSE ----
    const int idx = idxLds[t];
    const float* erow = emb + idx * 64;
    float acc = 0.f;
    #pragma unroll 8
    for (int d = 0; d < 64; ++d) {
        const float q  = erow[d];
        const float zv = zb[d * 4096 + t];
        ob[d * 4096 + t] = q;                 // straight-through forward value == zq
        const float df = q - zv;
        acc = __builtin_fmaf(df, df, acc);
    }
    #pragma unroll
    for (int mask = 16; mask >= 1; mask >>= 1) acc += __shfl_xor(acc, mask, 32);
    if (lane == 0) redLds[wv] = acc;
    __syncthreads();
    if (t == 0) {
        float s2 = 0.f;
        #pragma unroll
        for (int i = 0; i < 8; ++i) s2 += redLds[i];
        // loss = (1 + 0.25) * mean((zq - z)^2), mean over 32*64*64*64 elements
        partials[blockIdx.x] = s2 * (1.25f / 8388608.0f);
    }
}

// ---------------- Kernel 3: deterministic partial-sum reduction ----------------
__global__ void vq_loss(const float* __restrict__ partials, float* __restrict__ lossOut) {
    __shared__ float red[8];
    const int t = threadIdx.x;
    float a = partials[t] + partials[t + 256];
    #pragma unroll
    for (int mask = 16; mask >= 1; mask >>= 1) a += __shfl_xor(a, mask, 32);
    if ((t & 31) == 0) red[t >> 5] = a;
    __syncthreads();
    if (t == 0) {
        float s = 0.f;
        #pragma unroll
        for (int i = 0; i < 8; ++i) s += red[i];
        *lossOut = s;
    }
}

extern "C" void kernel_launch(void* const* d_in, const int* in_sizes, int n_in,
                              void* d_out, int out_size, void* d_ws, size_t ws_size,
                              hipStream_t stream) {
    (void)in_sizes; (void)n_in; (void)out_size; (void)ws_size;
    const float* z   = (const float*)d_in[0];   // [32,64,64,64]
    const float* emb = (const float*)d_in[1];   // [1024,64]
    float* out = (float*)d_out;                 // [32,64,64,64] then loss scalar

    char* ws = (char*)d_ws;
    _Float16* e16   = (_Float16*)ws;                         // 1024*64*2 = 131072 B
    float*    enorm = (float*)(ws + 131072);                 // 4096 B
    float*    parts = (float*)(ws + 131072 + 4096);          // 512*4 B

    vq_prep<<<1024, 64, 0, stream>>>(emb, e16, enorm);
    vq_main<<<512, 256, 0, stream>>>(z, emb, e16, enorm, out, parts);
    vq_loss<<<1, 256, 0, stream>>>(parts, out + 8388608);
}
